// TransformerLayer_10831907521041
// MI455X (gfx1250) — matrix operand order repaired and checked by reading, not compiled, not run on hardware
//
#include <hip/hip_runtime.h>
#include <hip/hip_bf16.h>

typedef __attribute__((ext_vector_type(16))) __bf16 v16bf;
typedef __attribute__((ext_vector_type(8)))  __bf16 v8bf;
typedef __attribute__((ext_vector_type(8)))  float  v8f;
typedef __attribute__((ext_vector_type(4)))  unsigned int v4u;
typedef __attribute__((ext_vector_type(8)))  int v8i;
typedef __attribute__((ext_vector_type(4)))  int v4i;

#if defined(__HIP_DEVICE_COMPILE__)
#if __has_builtin(__builtin_amdgcn_tensor_load_to_lds)
#define USE_TDM 1
#else
#define USE_TDM 0
#endif
#if __has_builtin(__builtin_amdgcn_permlane16)
#define USE_PERMLANE 1
#else
#define USE_PERMLANE 0
#endif
#else
#define USE_TDM 0
#define USE_PERMLANE 0
#endif

// ---------------------------------------------------------------------------
// WMMA helpers (CDNA5: D = A(16x32 bf16) * B(32x16 bf16) + C(16x16 f32))
// ---------------------------------------------------------------------------
__device__ __forceinline__ v8f wmma_bf16(v16bf a, v16bf b, v8f c) {
  return __builtin_amdgcn_wmma_f32_16x16x32_bf16(false, a, false, b, (short)0, c,
                                                 false, false);
}

// A-fragment: lane half selects K-base 0/8; elems 0..7 = K kb..kb+7,
// elems 8..15 = K kb+16..kb+23  -> two 16B loads at p and p+16 elements.
__device__ __forceinline__ v16bf frag_ld_a(const __bf16* p) {
  union { v16bf v; v8bf h[2]; } u;
  u.h[0] = *(const v8bf*)(p);
  u.h[1] = *(const v8bf*)(p + 16);
  return u.v;
}
// B-fragment (from K-contiguous storage of B^T): caller passes base + half*16;
// same K order as A: elems 0..7 = K 8*half.., elems 8..15 = K 16+8*half..
__device__ __forceinline__ v16bf frag_ld_b(const __bf16* p) {
  union { v16bf v; v8bf h[2]; } u;
  const int hb = ((threadIdx.x & 31) >> 4) * 8;
  u.h[0] = *(const v8bf*)(p - hb);
  u.h[1] = *(const v8bf*)(p - hb + 16);
  return u.v;
}

__device__ __forceinline__ v16bf ones16() {
  v16bf v;
#pragma unroll
  for (int i = 0; i < 16; ++i) v[i] = (__bf16)1.0f;
  return v;
}

// ---------------------------------------------------------------------------
// 16-lane-row max butterfly: v_permlane16_b32 (VALU) when available
// ---------------------------------------------------------------------------
#if USE_PERMLANE
__device__ __forceinline__ float permlanef(float x, unsigned lo, unsigned hi) {
  int xi = __builtin_bit_cast(int, x);
  int r = __builtin_amdgcn_permlane16(xi, xi, (int)lo, (int)hi, false, false);
  return __builtin_bit_cast(float, r);
}
#endif
__device__ __forceinline__ float rowmax16(float x) {
#if USE_PERMLANE
  x = fmaxf(x, permlanef(x, 0x67452301u, 0xEFCDAB89u));  // xor 1
  x = fmaxf(x, permlanef(x, 0x54761032u, 0xDCFE98BAu));  // xor 2
  x = fmaxf(x, permlanef(x, 0x32107654u, 0xBA98FEDCu));  // xor 4
  x = fmaxf(x, permlanef(x, 0xFEDCBA98u, 0x76543210u));  // xor 8
#else
  for (int off = 8; off; off >>= 1) x = fmaxf(x, __shfl_xor(x, off, 32));
#endif
  return x;
}

// ---------------------------------------------------------------------------
// Tensor Data Mover: 2D tile (global -> LDS) with LDS padding.
// D# layout per CDNA5 ISA.
// ---------------------------------------------------------------------------
#if USE_TDM
__device__ __forceinline__ void tdm_load_tile_2d(
    const void* gaddr, unsigned lds_off, unsigned tensor_d0, unsigned tensor_d1,
    unsigned d0_stride, unsigned tile_d0, unsigned tile_d1,
    unsigned pad_interval_code, unsigned pad_amount_code) {
  unsigned long long ga = (unsigned long long)(uintptr_t)gaddr;
  v4u g0;
  g0[0] = 1u;                                                   // count=1, no gather
  g0[1] = lds_off;                                              // lds_addr
  g0[2] = (unsigned)ga;                                         // global_addr[31:0]
  g0[3] = (unsigned)((ga >> 32) & 0x01FFFFFFull) | (2u << 30);  // addr[56:32] | type=2
  v8i g1;
  g1[0] = (int)((1u << 16) |                // data_size = 2 bytes
                (1u << 20) |                // pad_enable
                (pad_interval_code << 22) | (pad_amount_code << 25));
  g1[1] = (int)((tensor_d0 & 0xFFFFu) << 16);
  g1[2] = (int)(((tensor_d0 >> 16) & 0xFFFFu) | ((tensor_d1 & 0xFFFFu) << 16));
  g1[3] = (int)(((tensor_d1 >> 16) & 0xFFFFu) | (tile_d0 << 16));
  g1[4] = (int)(tile_d1 & 0xFFFFu);         // tile_dim1 | tile_dim2=0
  g1[5] = (int)d0_stride;                   // tensor_dim0_stride[31:0]
  g1[6] = 0;
  g1[7] = 0;
  v4i z4{};
#if __clang_major__ >= 23
  v8i z8{};
  __builtin_amdgcn_tensor_load_to_lds(g0, g1, z4, z4, z8, 0);
#else
  __builtin_amdgcn_tensor_load_to_lds(g0, g1, z4, z4, 0);
#endif
}
#endif

// ---------------------------------------------------------------------------
// fp32 -> bf16 conversion
// ---------------------------------------------------------------------------
__global__ __launch_bounds__(256) void cvt_f32_bf16(const float* __restrict__ in,
                                                    __bf16* __restrict__ out, int n) {
  int i = blockIdx.x * 256 + threadIdx.x;
  if (i < n) out[i] = (__bf16)in[i];
}

// ---------------------------------------------------------------------------
// LayerNorm (one block per row of D=1024), fp32 in -> bf16 out
// ---------------------------------------------------------------------------
__global__ __launch_bounds__(256) void ln_bf16(const float* __restrict__ x,
                                               const float* __restrict__ g,
                                               const float* __restrict__ b,
                                               __bf16* __restrict__ y, int D) {
  __shared__ float sm[8], sv[8];
  const int r = blockIdx.x;
  const float* xr = x + (size_t)r * D;
  float s = 0.f, s2 = 0.f;
  for (int i = threadIdx.x; i < D; i += 256) { float v = xr[i]; s += v; s2 += v * v; }
  for (int off = 16; off; off >>= 1) {
    s  += __shfl_xor(s, off, 32);
    s2 += __shfl_xor(s2, off, 32);
  }
  const int w = threadIdx.x >> 5, lane = threadIdx.x & 31;
  if (lane == 0) { sm[w] = s; sv[w] = s2; }
  __syncthreads();
  if (w == 0) {
    s  = (lane < 8) ? sm[lane] : 0.f;
    s2 = (lane < 8) ? sv[lane] : 0.f;
    for (int off = 4; off; off >>= 1) {
      s  += __shfl_xor(s, off, 32);
      s2 += __shfl_xor(s2, off, 32);
    }
    if (lane == 0) { sm[0] = s; sv[0] = s2; }
  }
  __syncthreads();
  const float mu  = sm[0] / (float)D;
  const float var = sv[0] / (float)D - mu * mu;
  const float inv = rsqrtf(var + 1e-5f);
  for (int i = threadIdx.x; i < D; i += 256)
    y[(size_t)r * D + i] = (__bf16)((xr[i] - mu) * inv * g[i] + b[i]);
}

// ---------------------------------------------------------------------------
// bf16 WMMA GEMM: C[M,N] = A[M,K] (row-major) x B[K,N] (row-major)
// 128x128 block tile, 8 waves, each wave 4x2 of 16x16 WMMA tiles, K step 32.
// A tile: TDM DMA (wave 0) with LDS padding; B tile: register transpose.
// Fused epilogue: +bias, relu, +residual, f32 or bf16 output.
// ---------------------------------------------------------------------------
template <bool OUT_BF16, bool BIAS, bool RELU, bool RES>
__global__ __launch_bounds__(256) void gemm_wmma(
    const __bf16* __restrict__ A, const __bf16* __restrict__ Bm,
    const float* __restrict__ bias, const float* __restrict__ res,
    void* __restrict__ Cout, int M, int N, int K) {
  __shared__ __bf16 All[128 * 40];  // A tile, row-major, stride 40 (16B aligned, conflict-free)
  __shared__ __bf16 Btl[128 * 40];  // B tile transposed: Bt[n][k], stride 40
  (void)M;

  const int t = threadIdx.x;
  const int lane = t & 31, w = t >> 5;
  const int lm = lane & 15, hi = lane >> 4;
  const int wm = w >> 2, wn = w & 3;
  const int rowbase = blockIdx.y * 128, colbase = blockIdx.x * 128;

#if !USE_TDM
  const int arow = t >> 1, akh = (t & 1) * 16;
#endif
  const int bk0 = w * 4, bn0 = lane * 4;

  v8f acc[4][2] = {};

  for (int k0 = 0; k0 < K; k0 += 32) {
    // --- stage A tile (128 rows x 32 K, padded to stride 40 halfwords) ---
#if USE_TDM
    if (w == 0) {
      // pad_interval: 16 DWORDs (code 3) = one 32-elem bf16 row; pad: 4 DWORDs (code 3)
      tdm_load_tile_2d(A + (size_t)rowbase * K + k0, (unsigned)(uintptr_t)All,
                       (unsigned)K, (unsigned)M, (unsigned)K, 32u, 128u, 3u, 3u);
    }
#else
    {
      const __bf16* ag = A + (size_t)(rowbase + arow) * K + k0 + akh;
      *(v8bf*)&All[arow * 40 + akh]     = *(const v8bf*)(ag);
      *(v8bf*)&All[arow * 40 + akh + 8] = *(const v8bf*)(ag + 8);
    }
#endif
    // --- stage B tile, transposed in registers ---
    const unsigned short* bg =
        (const unsigned short*)Bm + (size_t)(k0 + bk0) * N + colbase + bn0;
    ushort4 r0 = *(const ushort4*)(bg);
    ushort4 r1 = *(const ushort4*)(bg + (size_t)N);
    ushort4 r2 = *(const ushort4*)(bg + (size_t)2 * N);
    ushort4 r3 = *(const ushort4*)(bg + (size_t)3 * N);
    if (k0 + 32 < K) __builtin_prefetch((const void*)(bg + (size_t)32 * N), 0, 0);
    unsigned short* btp = (unsigned short*)Btl;
    *(ushort4*)(btp + (bn0 + 0) * 40 + bk0) = make_ushort4(r0.x, r1.x, r2.x, r3.x);
    *(ushort4*)(btp + (bn0 + 1) * 40 + bk0) = make_ushort4(r0.y, r1.y, r2.y, r3.y);
    *(ushort4*)(btp + (bn0 + 2) * 40 + bk0) = make_ushort4(r0.z, r1.z, r2.z, r3.z);
    *(ushort4*)(btp + (bn0 + 3) * 40 + bk0) = make_ushort4(r0.w, r1.w, r2.w, r3.w);
#if USE_TDM
    if (w == 0) __builtin_amdgcn_s_wait_tensorcnt(0);
#endif
    __syncthreads();

    v16bf bf0 = frag_ld_b(&Btl[(wn * 32 + lm) * 40 + hi * 16]);
    v16bf bf1 = frag_ld_b(&Btl[(wn * 32 + 16 + lm) * 40 + hi * 16]);
#pragma unroll
    for (int ms = 0; ms < 4; ++ms) {
      v16bf af = frag_ld_a(&All[(wm * 64 + ms * 16 + lm) * 40 + hi * 8]);
      acc[ms][0] = wmma_bf16(af, bf0, acc[ms][0]);
      acc[ms][1] = wmma_bf16(af, bf1, acc[ms][1]);
    }
    __syncthreads();
  }

  // --- epilogue: C layout row = hi*8 + vgpr, col = lm ---
#pragma unroll
  for (int ms = 0; ms < 4; ++ms)
#pragma unroll
    for (int ns = 0; ns < 2; ++ns) {
      const int rr = rowbase + wm * 64 + ms * 16 + hi * 8;
      const int c  = colbase + wn * 32 + ns * 16 + lm;
#pragma unroll
      for (int i = 0; i < 8; ++i) {
        const int r = rr + i;
        float vv = acc[ms][ns][i];
        if (BIAS) vv += bias[c];
        if (RELU) vv = vv > 0.f ? vv : 0.f;
        if (RES)  vv += res[(size_t)r * N + c];
        if (OUT_BF16) ((__bf16*)Cout)[(size_t)r * N + c] = (__bf16)vv;
        else          ((float*)Cout)[(size_t)r * N + c]  = vv;
      }
    }
}

// ---------------------------------------------------------------------------
// Flash attention (non-causal), bf16 q/k/v -> bf16 ctx. DH=64, key chunks of 64.
// Each block: 8 waves x 16 queries = 128 queries for one (b,h).
// K chunk: TDM DMA; V chunk: register transpose; row-sums via WMMA vs ones.
// ---------------------------------------------------------------------------
__global__ __launch_bounds__(256) void flash_attn(const __bf16* __restrict__ q,
                                                  const __bf16* __restrict__ k,
                                                  const __bf16* __restrict__ v,
                                                  __bf16* __restrict__ ctx) {
  constexpr int S = 2048, Bb = 2, D = 1024, DH = 64;
  __shared__ __bf16 Kl[64 * 72];      // K chunk row-major [key][dim]  (== Bt for QK^T)
  __shared__ __bf16 Vt[64 * 72];      // V chunk transposed [dim][key] (== Bt for PV)
  __shared__ __bf16 Pl[8][16 * 72];   // per-wave P staging (C-layout -> A-layout)

  const int t = threadIdx.x, lane = t & 31, w = t >> 5;
  const int lm = lane & 15, hi = lane >> 4;
  const int bh = blockIdx.y;
  const int b = bh >> 4, h = bh & 15;  // H = 16
  const int qbase = blockIdx.x * 128 + w * 16;

  // Q fragments live in registers (A layout, gathered straight from global)
  const __bf16* qrow = q + ((size_t)(qbase + lm) * Bb + b) * D + h * DH;
  v16bf qf[2];
  qf[0] = frag_ld_a(qrow + hi * 8);
  qf[1] = frag_ld_a(qrow + 32 + hi * 8);
  const v16bf onesf = ones16();

  float mrow[8], lrow[8], scale8[8];
  v8f o[4] = {};
#pragma unroll
  for (int i = 0; i < 8; ++i) { mrow[i] = -1e30f; lrow[i] = 0.f; }

#if !USE_TDM
  const int kkey = t >> 2, kdh = (t & 3) * 16;               // K: 64 keys x 4 halves
#endif
  const int vd0 = (t & 15) * 4, vk0 = (t >> 4) * 4;          // V: 4dim x 4key transpose

  for (int tc = 0; tc < S; tc += 64) {
    // stage K chunk
#if USE_TDM
    if (w == 0) {
      // pad_interval: 32 DWORDs (code 4) = one 64-elem bf16 row; pad: 4 DWORDs (code 3)
      tdm_load_tile_2d(k + ((size_t)tc * Bb + b) * D + h * DH, (unsigned)(uintptr_t)Kl,
                       (unsigned)(Bb * D), (unsigned)S, (unsigned)(Bb * D), 64u, 64u,
                       4u, 3u);
    }
#else
    {
      const __bf16* kg = k + ((size_t)(tc + kkey) * Bb + b) * D + h * DH + kdh;
      *(v8bf*)&Kl[kkey * 72 + kdh]     = *(const v8bf*)(kg);
      *(v8bf*)&Kl[kkey * 72 + kdh + 8] = *(const v8bf*)(kg + 8);
    }
#endif
    // stage V chunk transposed
    {
      const unsigned short* vg =
          (const unsigned short*)v + ((size_t)(tc + vk0) * Bb + b) * D + h * DH + vd0;
      const size_t rstride = (size_t)Bb * D;
      ushort4 r0 = *(const ushort4*)(vg);
      ushort4 r1 = *(const ushort4*)(vg + rstride);
      ushort4 r2 = *(const ushort4*)(vg + 2 * rstride);
      ushort4 r3 = *(const ushort4*)(vg + 3 * rstride);
      if (tc + 64 < S) __builtin_prefetch((const void*)(vg + 64 * rstride), 0, 0);
      unsigned short* vt = (unsigned short*)Vt;
      *(ushort4*)(vt + (vd0 + 0) * 72 + vk0) = make_ushort4(r0.x, r1.x, r2.x, r3.x);
      *(ushort4*)(vt + (vd0 + 1) * 72 + vk0) = make_ushort4(r0.y, r1.y, r2.y, r3.y);
      *(ushort4*)(vt + (vd0 + 2) * 72 + vk0) = make_ushort4(r0.z, r1.z, r2.z, r3.z);
      *(ushort4*)(vt + (vd0 + 3) * 72 + vk0) = make_ushort4(r0.w, r1.w, r2.w, r3.w);
    }
#if USE_TDM
    if (w == 0) __builtin_amdgcn_s_wait_tensorcnt(0);
#endif
    __syncthreads();

    // scores: S_tile[16 x 64] = Q[16x64] * Kchunk^T, 4 column tiles, K split 2x32
    v8f sc[4];
#pragma unroll
    for (int nt = 0; nt < 4; ++nt) {
      v8f a = {};
#pragma unroll
      for (int kc = 0; kc < 2; ++kc) {
        v16bf bfr = frag_ld_b(&Kl[(nt * 16 + lm) * 72 + kc * 32 + hi * 16]);
        a = wmma_bf16(qf[kc], bfr, a);
      }
      sc[nt] = a * 0.125f;  // 1/sqrt(DH)
    }

    // row max (VALU permlane butterfly) + exp
#pragma unroll
    for (int i = 0; i < 8; ++i) {
      float mx = fmaxf(fmaxf(sc[0][i], sc[1][i]), fmaxf(sc[2][i], sc[3][i]));
      mx = rowmax16(mx);
      const float nm = fmaxf(mrow[i], mx);
      scale8[i] = __expf(mrow[i] - nm);
      mrow[i] = nm;
#pragma unroll
      for (int nt = 0; nt < 4; ++nt) sc[nt][i] = __expf(sc[nt][i] - nm);
    }

    // C-layout -> A-layout via wave-private LDS
    __bf16* pw = &Pl[w][0];
#pragma unroll
    for (int nt = 0; nt < 4; ++nt)
#pragma unroll
      for (int i = 0; i < 8; ++i)
        pw[(hi * 8 + i) * 72 + nt * 16 + lm] = (__bf16)sc[nt][i];
    asm volatile("s_wait_dscnt 0x0" ::: "memory");  // wave-synchronous LDS reuse
    v16bf pf0 = frag_ld_a(&pw[lm * 72 + hi * 8]);
    v16bf pf1 = frag_ld_a(&pw[lm * 72 + 32 + hi * 8]);

    // row sums via WMMA against ones (C layout matches lrow indexing exactly)
    v8f rs = {};
    rs = wmma_bf16(pf0, onesf, rs);
    rs = wmma_bf16(pf1, onesf, rs);

#pragma unroll
    for (int i = 0; i < 8; ++i) {
      lrow[i] = lrow[i] * scale8[i] + rs[i];
#pragma unroll
      for (int dt = 0; dt < 4; ++dt) o[dt][i] *= scale8[i];
    }

    // O += P[16x64] * Vchunk[64x64]
#pragma unroll
    for (int dt = 0; dt < 4; ++dt) {
      v16bf b0 = frag_ld_b(&Vt[(dt * 16 + lm) * 72 + hi * 16]);
      v16bf b1 = frag_ld_b(&Vt[(dt * 16 + lm) * 72 + 32 + hi * 16]);
      o[dt] = wmma_bf16(pf0, b0, o[dt]);
      o[dt] = wmma_bf16(pf1, b1, o[dt]);
    }
    __syncthreads();
  }

  // normalize + store ctx (bf16)
#pragma unroll
  for (int dt = 0; dt < 4; ++dt)
#pragma unroll
    for (int i = 0; i < 8; ++i) {
      const int srow = qbase + hi * 8 + i;
      const float ov = o[dt][i] / lrow[i];
      ctx[((size_t)srow * Bb + b) * D + h * DH + dt * 16 + lm] = (__bf16)ov;
    }
}

// ---------------------------------------------------------------------------
// Orchestration
// ---------------------------------------------------------------------------
extern "C" void kernel_launch(void* const* d_in, const int* in_sizes, int n_in,
                              void* d_out, int out_size, void* d_ws, size_t ws_size,
                              hipStream_t stream) {
  (void)in_sizes; (void)n_in; (void)out_size; (void)ws_size;
  constexpr int S = 2048, B = 2, D = 1024, H = 16, FF = 4096;
  constexpr int R = S * B;  // 4096 rows

  const float* x     = (const float*)d_in[0];
  const float* wq    = (const float*)d_in[1];
  const float* wk    = (const float*)d_in[2];
  const float* wv    = (const float*)d_in[3];
  const float* wo    = (const float*)d_in[4];
  const float* ln1_g = (const float*)d_in[5];
  const float* ln1_b = (const float*)d_in[6];
  const float* ln2_g = (const float*)d_in[7];
  const float* ln2_b = (const float*)d_in[8];
  const float* w1    = (const float*)d_in[9];
  const float* b1    = (const float*)d_in[10];
  const float* w2    = (const float*)d_in[11];
  const float* b2    = (const float*)d_in[12];

  char* wsp = (char*)d_ws;
  auto carve = [&](size_t bytes) -> char* {
    char* p = wsp;
    wsp += (bytes + 255) & ~(size_t)255;
    return p;
  };
  __bf16* wq_b  = (__bf16*)carve((size_t)D * D * 2);
  __bf16* wk_b  = (__bf16*)carve((size_t)D * D * 2);
  __bf16* wv_b  = (__bf16*)carve((size_t)D * D * 2);
  __bf16* wo_b  = (__bf16*)carve((size_t)D * D * 2);
  __bf16* w1_b  = (__bf16*)carve((size_t)D * FF * 2);
  __bf16* w2_b  = (__bf16*)carve((size_t)FF * D * 2);
  __bf16* y_b   = (__bf16*)carve((size_t)R * D * 2);
  __bf16* q_b   = (__bf16*)carve((size_t)R * D * 2);
  __bf16* k_b   = (__bf16*)carve((size_t)R * D * 2);
  __bf16* v_b   = (__bf16*)carve((size_t)R * D * 2);
  __bf16* ctx_b = (__bf16*)carve((size_t)R * D * 2);
  float*  x1    = (float*) carve((size_t)R * D * 4);
  __bf16* z_b   = (__bf16*)carve((size_t)R * D * 2);
  __bf16* h_b   = (__bf16*)carve((size_t)R * FF * 2);

  // weights -> bf16
  cvt_f32_bf16<<<(D * D + 255) / 256, 256, 0, stream>>>(wq, wq_b, D * D);
  cvt_f32_bf16<<<(D * D + 255) / 256, 256, 0, stream>>>(wk, wk_b, D * D);
  cvt_f32_bf16<<<(D * D + 255) / 256, 256, 0, stream>>>(wv, wv_b, D * D);
  cvt_f32_bf16<<<(D * D + 255) / 256, 256, 0, stream>>>(wo, wo_b, D * D);
  cvt_f32_bf16<<<(D * FF + 255) / 256, 256, 0, stream>>>(w1, w1_b, D * FF);
  cvt_f32_bf16<<<(FF * D + 255) / 256, 256, 0, stream>>>(w2, w2_b, FF * D);

  // sublayer 0: attention
  ln_bf16<<<R, 256, 0, stream>>>(x, ln1_g, ln1_b, y_b, D);
  dim3 gDD(D / 128, R / 128);
  gemm_wmma<true, false, false, false><<<gDD, 256, 0, stream>>>(
      y_b, wq_b, nullptr, nullptr, q_b, R, D, D);
  gemm_wmma<true, false, false, false><<<gDD, 256, 0, stream>>>(
      y_b, wk_b, nullptr, nullptr, k_b, R, D, D);
  gemm_wmma<true, false, false, false><<<gDD, 256, 0, stream>>>(
      y_b, wv_b, nullptr, nullptr, v_b, R, D, D);
  dim3 gAt(S / 128, B * H);
  flash_attn<<<gAt, 256, 0, stream>>>(q_b, k_b, v_b, ctx_b);
  gemm_wmma<false, false, false, true><<<gDD, 256, 0, stream>>>(
      ctx_b, wo_b, nullptr, x, x1, R, D, D);

  // sublayer 1: FFN
  ln_bf16<<<R, 256, 0, stream>>>(x1, ln2_g, ln2_b, z_b, D);
  dim3 gFF(FF / 128, R / 128);
  gemm_wmma<true, true, true, false><<<gFF, 256, 0, stream>>>(
      z_b, w1_b, b1, nullptr, h_b, R, FF, D);
  dim3 gD2(D / 128, R / 128);
  gemm_wmma<false, true, false, true><<<gD2, 256, 0, stream>>>(
      h_b, w2_b, b2, x1, (float*)d_out, R, D, FF);
}
